// Module_523986010721
// MI455X (gfx1250) — compile-verified, matrix operand order checked by reading, not run on hardware
//
#include <hip/hip_runtime.h>
#include <hip/hip_bf16.h>
#include <math.h>

typedef __attribute__((ext_vector_type(16))) __bf16 v16bf;
typedef __attribute__((ext_vector_type(8)))  __bf16 v8bf;
typedef __attribute__((ext_vector_type(4)))  __bf16 v4bf;
typedef __attribute__((ext_vector_type(8)))  float  v8f;
typedef __attribute__((ext_vector_type(4)))  float  v4f;
typedef int b128i __attribute__((vector_size(16)));   // matches builtin param type

#define N_CELLS 2048
#define CDIM    256
#define NFUNC   24
#define NARG    16
#define EDIM    128
#define GEXP    6

// --------------------------- CDNA5 async Global->LDS ------------------------
#if __has_builtin(__builtin_amdgcn_global_load_async_to_lds_b128)
#define USE_ASYNC_LDS 1
#else
#define USE_ASYNC_LDS 0
#endif

__device__ __forceinline__ void async_wait_all() {
#if USE_ASYNC_LDS
#if __has_builtin(__builtin_amdgcn_s_wait_asynccnt)
    __builtin_amdgcn_s_wait_asynccnt(0);
#else
    asm volatile("s_wait_asynccnt 0x0" ::: "memory");
#endif
#endif
}

#if USE_ASYNC_LDS
__device__ __forceinline__ void async_cp16(const __bf16* g, __bf16* l) {
    __builtin_amdgcn_global_load_async_to_lds_b128(
        (__attribute__((address_space(1))) b128i*)(uintptr_t)g,
        (__attribute__((address_space(3))) b128i*)(uint32_t)(uintptr_t)l,
        0, 0);
}
#endif

// ---------------------------------------------------------------------------
// Kernel 1: softmax(p_func[1:]) -> pf [N,24];  softmax(p_args) -> pa;
//           xr[n,r,:] = sum_k pa[n,r,k] * args[n,k,:]   (bf16, [N,3,128])
// ---------------------------------------------------------------------------
__global__ void __launch_bounds__(128) prep_kernel(
    const float* __restrict__ p_func, const float* __restrict__ args,
    const float* __restrict__ p_args, float* __restrict__ pf,
    __bf16* __restrict__ xr)
{
    const int n = blockIdx.x;
    const int t = threadIdx.x; // 128 threads
    __shared__ float pa_s[3][NARG];
    __shared__ float argss[NARG][EDIM];

    for (int k = 0; k < NARG; ++k)
        argss[k][t] = args[((size_t)n * NARG + k) * EDIM + t];

    if (t < 3) { // per-arity softmax over 16 candidates
        const float* row = p_args + ((size_t)n * 3 + t) * NARG;
        float m = -1e30f;
        for (int k = 0; k < NARG; ++k) m = fmaxf(m, row[k]);
        float s = 0.f;
        for (int k = 0; k < NARG; ++k) { float e = __expf(row[k] - m); pa_s[t][k] = e; s += e; }
        float inv = 1.f / s;
        for (int k = 0; k < NARG; ++k) pa_s[t][k] *= inv;
    }
    if (t == 3) { // function softmax over indices 1..23, index 0 -> 0
        const float* row = p_func + (size_t)n * NFUNC;
        float m = -1e30f;
        for (int i = 1; i < NFUNC; ++i) m = fmaxf(m, row[i]);
        float s = 0.f, e[NFUNC];
        for (int i = 1; i < NFUNC; ++i) { e[i] = __expf(row[i] - m); s += e[i]; }
        float inv = 1.f / s;
        pf[(size_t)n * NFUNC] = 0.f;
        for (int i = 1; i < NFUNC; ++i) pf[(size_t)n * NFUNC + i] = e[i] * inv;
    }
    __syncthreads();

    for (int r = 0; r < 3; ++r) {
        float s = 0.f;
        for (int k = 0; k < NARG; ++k) s += pa_s[r][k] * argss[k][t];
        xr[((size_t)n * 3 + r) * EDIM + t] = (__bf16)s;
    }
}

// ---------------------------------------------------------------------------
// Kernel 2: out_acc[n,c] = sum_g pf[n, 4g] * const0[g,c]   (arity-0 experts)
// ---------------------------------------------------------------------------
__global__ void __launch_bounds__(256) init_out_kernel(
    const float* __restrict__ pf, const float* __restrict__ const0,
    float* __restrict__ out_acc)
{
    int idx = blockIdx.x * 256 + threadIdx.x;           // over N*C
    int n = idx >> 8, c = idx & 255;
    float s = 0.f;
    for (int g = 0; g < GEXP; ++g)
        s += pf[(size_t)n * NFUNC + g * 4] * const0[g * CDIM + c];
    out_acc[idx] = s;
}

// ---------------------------------------------------------------------------
// Kernel 3: f32 -> bf16 elementwise
// ---------------------------------------------------------------------------
__global__ void __launch_bounds__(256) cvt_bf16_kernel(
    const float* __restrict__ src, __bf16* __restrict__ dst, int n)
{
    int i = blockIdx.x * 256 + threadIdx.x;
    if (i < n) dst[i] = (__bf16)src[i];
}

// ---------------------------------------------------------------------------
// Generic bf16 WMMA GEMM:  out = act( A[MxK](bf16) * B[KxN](f32->bf16) + bias )
// Block tile 128x128, K-step 32, 8 waves (wave tile 32x64, 2x4 fragments),
// double-buffered LDS, async global->LDS for the A stream when available.
// mode 0: store bf16    mode 1: store f32
// mode 2: outF32[m,n] += scale[m*sstride] * v   (per-row expert weight)
// ---------------------------------------------------------------------------
__global__ void __launch_bounds__(256) gemm_bf16_wmma(
    const __bf16* __restrict__ A, int lda,
    const float* __restrict__ B, int ldb,
    const float* __restrict__ bias,
    int K, int act, int mode,
    void* __restrict__ out, int ldo,
    const float* __restrict__ scale, int sstride)
{
    __shared__ __align__(16) __bf16 As[2][128][48]; // 32 cols + pad, double buffered
    __shared__ __align__(16) __bf16 Bs[2][128][48]; // transposed [n][k] + pad

    const int t    = threadIdx.x;
    const int lane = t & 31;
    const int w    = t >> 5;
    const int waveM = (w & 3) * 32;
    const int waveN = (w >> 2) * 64;
    const int bm = blockIdx.y * 128;
    const int bn = blockIdx.x * 128;

    v8f acc[2][4];
    const v8f vzero = {0.f,0.f,0.f,0.f,0.f,0.f,0.f,0.f};
    for (int mi = 0; mi < 2; ++mi)
        for (int ni = 0; ni < 4; ++ni) acc[mi][ni] = vzero;

    const int r16 = lane & 15;          // fragment row/col within 16
    const int kb  = (lane >> 4) << 3;   // 0 or 8 : ISA 16-bit A/B K layout

    // ---- hoisted per-thread addressing -------------------------------------
    // A: thread owns rows {t>>2, t>>2+64}, 16B chunk (t&3); tile step = 32 elems
    const int rowA  = t >> 2;
    const int partA = (t & 3) * 8;
    const __bf16* Abase = A + (size_t)(bm + rowA) * lda + partA;
    const size_t  AhalfStep = (size_t)64 * lda;
    // B: thread owns 4 consecutive rows {4q..4q+3}, fixed 4-col group (t&31)*4
    const int kq = (t >> 5) * 4;        // 0,4,...,28
    const int n4 = (t & 31) * 4;
    const float* Bbase = B + (size_t)kq * ldb + bn + n4;
    const size_t BtileStep = (size_t)32 * ldb; // per K-tile

    auto loadA = [&](int tile, int buf) {
        const __bf16* src = Abase + (size_t)tile * 32;
#if USE_ASYNC_LDS
        async_cp16(src, &As[buf][rowA][partA]);
        async_cp16(src + AhalfStep, &As[buf][rowA + 64][partA]);
#else
        *(v8bf*)&As[buf][rowA][partA]      = *(const v8bf*)src;
        *(v8bf*)&As[buf][rowA + 64][partA] = *(const v8bf*)(src + AhalfStep);
#endif
    };
    auto loadB = [&](int tile, int buf) {
        const float* src = Bbase + (size_t)tile * BtileStep;
        v4f tmp[4];
#pragma unroll
        for (int j = 0; j < 4; ++j)                 // issue all 4 loads first
            tmp[j] = *(const v4f*)(src + (size_t)j * ldb);
#pragma unroll
        for (int c = 0; c < 4; ++c) {               // pack 4 k-values -> b64 store
            v4bf p = { (__bf16)tmp[0][c], (__bf16)tmp[1][c],
                       (__bf16)tmp[2][c], (__bf16)tmp[3][c] };
            *(v4bf*)&Bs[buf][n4 + c][kq] = p;
        }
    };

    const int nk = K >> 5;
    loadA(0, 0);
    loadB(0, 0);
    async_wait_all();
    __syncthreads();

    for (int kt = 0; kt < nk; ++kt) {
        const int buf = kt & 1;
        if (kt + 1 < nk) {                       // prefetch next tile
            loadA(kt + 1, buf ^ 1);
            loadB(kt + 1, buf ^ 1);
        }

        v16bf af[2], bf[4];
#pragma unroll
        for (int mi = 0; mi < 2; ++mi) {
            int r = waveM + mi * 16 + r16;
            v8bf lo = *(const v8bf*)&As[buf][r][kb];
            v8bf hi = *(const v8bf*)&As[buf][r][kb + 16];
            af[mi] = __builtin_shufflevector(lo, hi, 0,1,2,3,4,5,6,7,8,9,10,11,12,13,14,15);
        }
#pragma unroll
        for (int ni = 0; ni < 4; ++ni) {
            int c = waveN + ni * 16 + r16;
            v8bf lo = *(const v8bf*)&Bs[buf][c][kb];
            v8bf hi = *(const v8bf*)&Bs[buf][c][kb + 16];
            bf[ni] = __builtin_shufflevector(lo, hi, 0,1,2,3,4,5,6,7,8,9,10,11,12,13,14,15);
        }
#pragma unroll
        for (int mi = 0; mi < 2; ++mi)
#pragma unroll
            for (int ni = 0; ni < 4; ++ni)
                acc[mi][ni] = __builtin_amdgcn_wmma_f32_16x16x32_bf16(
                    false, af[mi], false, bf[ni], (short)0, acc[mi][ni], false, false);

        async_wait_all();                        // async writes to buf^1 done
        __syncthreads();                         // all waves done reading buf
    }

    // --- epilogue: C/D layout -> (m = r + 8*(lane>=16), n = lane&15) ---
    const int mhi = (lane >> 4) * 8;
#pragma unroll
    for (int ni = 0; ni < 4; ++ni) {
        int col = bn + waveN + ni * 16 + r16;
        float bia = bias[col];
#pragma unroll
        for (int mi = 0; mi < 2; ++mi) {
            int rowbase = bm + waveM + mi * 16 + mhi;
#pragma unroll
            for (int r = 0; r < 8; ++r) {
                int row = rowbase + r;
                float v = acc[mi][ni][r] + bia;
                if (act) v = tanhf(v);
                if (mode == 0)
                    ((__bf16*)out)[(size_t)row * ldo + col] = (__bf16)v;
                else if (mode == 1)
                    ((float*)out)[(size_t)row * ldo + col] = v;
                else
                    ((float*)out)[(size_t)row * ldo + col] += scale[(size_t)row * sstride] * v;
            }
        }
    }
}

// ---------------------------------------------------------------------------
extern "C" void kernel_launch(void* const* d_in, const int* in_sizes, int n_in,
                              void* d_out, int out_size, void* d_ws, size_t ws_size,
                              hipStream_t stream) {
    const float* p_func = (const float*)d_in[0];
    const float* args_  = (const float*)d_in[1];
    const float* p_args = (const float*)d_in[2];
    const float* enc_W  = (const float*)d_in[3];
    const float* enc_b  = (const float*)d_in[4];
    const float* dec_W  = (const float*)d_in[5];
    const float* dec_b  = (const float*)d_in[6];
    const float* const0 = (const float*)d_in[7];

    // workspace carve-up (all 256B aligned)
    char* ws = (char*)d_ws;
    size_t off = 0;
    auto carve = [&](size_t bytes) { void* p = ws + off; off += (bytes + 255) & ~(size_t)255; return p; };
    float*  pf      = (float*) carve((size_t)N_CELLS * NFUNC * 4);
    __bf16* xr      = (__bf16*)carve((size_t)N_CELLS * 3 * EDIM * 2);
    __bf16* aggB    = (__bf16*)carve((size_t)N_CELLS * 3 * CDIM * 2);     // [N, 768]
    __bf16* T1      = (__bf16*)carve((size_t)N_CELLS * 12 * CDIM * 2);    // up to [N, 3072]
    __bf16* T2      = (__bf16*)carve((size_t)N_CELLS * 4 * CDIM * 2);     // [N, 1024]
    float*  out_acc = (float*) carve((size_t)N_CELLS * CDIM * 4);
    __bf16* outB    = (__bf16*)carve((size_t)N_CELLS * CDIM * 2);

    auto gemm = [&](const __bf16* A, int lda, const float* B, int ldb,
                    const float* bias, int M, int K, int Nc, int act, int mode,
                    void* out, int ldo, const float* scale, int sstride) {
        dim3 grid(Nc / 128, M / 128);
        gemm_bf16_wmma<<<grid, dim3(256), 0, stream>>>(
            A, lda, B, ldb, bias, K, act, mode, out, ldo, scale, sstride);
    };

    // 1) softmaxes + candidate aggregation in raw-arg space
    prep_kernel<<<dim3(N_CELLS), dim3(128), 0, stream>>>(p_func, args_, p_args, pf, xr);

    // 2) encoder on aggregated args: agg = xr @ enc_W + enc_b  -> [N,3,256] = [N,768]
    gemm(xr, EDIM, enc_W, CDIM, enc_b, N_CELLS * 3, EDIM, CDIM, /*act*/0, /*mode*/0,
         aggB, CDIM, nullptr, 0);

    // 3) arity-0 experts seed the accumulator
    init_out_kernel<<<dim3(N_CELLS * CDIM / 256), dim3(256), 0, stream>>>(pf, const0, out_acc);

    // 4) expert MLPs, arity 1..3, experts g=0..5 (sequential launches, buffers reused)
    for (int a = 1; a <= 3; ++a) {
        int base = 8 + (a - 1) * 6;
        const float* W1 = (const float*)d_in[base + 0];
        const float* b1 = (const float*)d_in[base + 1];
        const float* W2 = (const float*)d_in[base + 2];
        const float* b2 = (const float*)d_in[base + 3];
        const float* W3 = (const float*)d_in[base + 4];
        const float* b3 = (const float*)d_in[base + 5];
        const int aC = a * CDIM;       // input width (prefix of aggB)
        const int H  = 4 * aC;         // hidden width
        for (int g = 0; g < GEXP; ++g) {
            // layer1: [N, aC] x [aC, H] -> tanh -> T1 (bf16)
            gemm(aggB, 3 * CDIM, W1 + (size_t)g * aC * H, H, b1 + (size_t)g * H,
                 N_CELLS, aC, H, 1, 0, T1, H, nullptr, 0);
            // layer2: [N, H] x [H, 4C] -> tanh -> T2 (bf16)
            gemm(T1, H, W2 + (size_t)g * H * (4 * CDIM), 4 * CDIM, b2 + (size_t)g * 4 * CDIM,
                 N_CELLS, H, 4 * CDIM, 1, 0, T2, 4 * CDIM, nullptr, 0);
            // layer3: [N, 4C] x [4C, C], scaled accumulate with pf[:, g*4+a]
            gemm(T2, 4 * CDIM, W3 + (size_t)g * 4 * CDIM * CDIM, CDIM, b3 + (size_t)g * CDIM,
                 N_CELLS, 4 * CDIM, CDIM, 0, 2, out_acc, CDIM,
                 pf + (g * 4 + a), NFUNC);
        }
    }

    // 5) decoder: (out_acc -> bf16) @ dec_W + dec_b -> d_out (f32)
    cvt_bf16_kernel<<<dim3(N_CELLS * CDIM / 256), dim3(256), 0, stream>>>(
        out_acc, outB, N_CELLS * CDIM);
    gemm(outB, CDIM, dec_W, CDIM, dec_b, N_CELLS, CDIM, CDIM, 0, 1, d_out, CDIM, nullptr, 0);
}
